// GraphAttentionLayer_35983236006299
// MI455X (gfx1250) — matrix operand order repaired and checked by reading, not compiled, not run on hardware
//
#include <hip/hip_runtime.h>

typedef __attribute__((ext_vector_type(16))) _Float16 v16h;
typedef __attribute__((ext_vector_type(8)))  _Float16 v8h;
typedef __attribute__((ext_vector_type(8)))  float    v8f;
typedef __attribute__((ext_vector_type(4)))  unsigned int v4u;
typedef __attribute__((ext_vector_type(4)))  int      v4i;
typedef __attribute__((ext_vector_type(8)))  int      v8i;

#define ALPHA_SLOPE 0.01f
#define NEG_INF_F  -1.0e12f

// ---------------------------------------------------------------------------
// Phase 1: h = inp @ W  (M=16384, N=64, K=128) via v_wmma_f32_16x16x32_f16,
// plus s1 = h@a1, s2 = h@a2, and store h transposed as f16: hT[b][n][j].
// One wave = one 16-row tile. 8 waves / block, 128 blocks.
// ---------------------------------------------------------------------------
__global__ __launch_bounds__(256) void gat_phase1(
    const float* __restrict__ inp, const float* __restrict__ W,
    const float* __restrict__ a,
    _Float16* __restrict__ hT, float* __restrict__ s1, float* __restrict__ s2)
{
    // W transposed in LDS: WT[n][k], stride 136 halves (272B rows: 16B-aligned
    // fragment loads; banks 68n mod 64 = 4n -> conflict-free over 16 lanes)
    __shared__ _Float16 ldsWT[64 * 136];
    const int tid = threadIdx.x;
    for (int idx = tid; idx < 128 * 64; idx += 256) {
        int k = idx >> 6, n = idx & 63;             // W row-major [k][n]
        ldsWT[n * 136 + k] = (_Float16)W[idx];
    }
    __syncthreads();

    const int wave = tid >> 5;
    const int lane = tid & 31;
    const int m    = lane & 15;
    const int half = lane >> 4;

    const int  rowTile = blockIdx.x * 8 + wave;     // 0..1023
    const long g0      = (long)rowTile * 16;        // first flat row (b*2048+j)
    const float* arow  = inp + (g0 + m) * 128;

    v8f acc[4] = {};

    #pragma unroll
    for (int ks = 0; ks < 4; ++ks) {
        const int k0 = ks * 32;
        // A fragment (16x32 f16): two contiguous 8-float runs per lane
        v16h af;
        #pragma unroll
        for (int seg = 0; seg < 2; ++seg) {
            const float* p = arow + k0 + 16 * seg + 8 * half;
            float4 u0 = *(const float4*)(p);
            float4 u1 = *(const float4*)(p + 4);
            af[8 * seg + 0] = (_Float16)u0.x;  af[8 * seg + 1] = (_Float16)u0.y;
            af[8 * seg + 2] = (_Float16)u0.z;  af[8 * seg + 3] = (_Float16)u0.w;
            af[8 * seg + 4] = (_Float16)u1.x;  af[8 * seg + 5] = (_Float16)u1.y;
            af[8 * seg + 6] = (_Float16)u1.z;  af[8 * seg + 7] = (_Float16)u1.w;
        }
        #pragma unroll
        for (int t = 0; t < 4; ++t) {
            const int n = t * 16 + m;
            // B fragment: two 8-half runs at [n][k0 + 8*half] and [n][k0 + 16 + 8*half]
            const int base = n * 136 + k0 + 8 * half;
            v8h lo = *(const v8h*)&ldsWT[base];
            v8h hi = *(const v8h*)&ldsWT[base + 16];
            v16h bf = __builtin_shufflevector(lo, hi,
                0,1,2,3,4,5,6,7,8,9,10,11,12,13,14,15);
            acc[t] = __builtin_amdgcn_wmma_f32_16x16x32_f16(
                false, af, false, bf, (short)0, acc[t], false, false);
        }
    }

    // s1/s2: acc[t][v] = h[row = v + 8*half][n = t*16 + m]
    #pragma unroll
    for (int v = 0; v < 8; ++v) {
        float p1 = 0.f, p2 = 0.f;
        #pragma unroll
        for (int t = 0; t < 4; ++t) {
            const int n = t * 16 + m;
            const float hv = acc[t][v];
            p1 += hv * a[n];
            p2 += hv * a[64 + n];
        }
        #pragma unroll
        for (int off = 1; off < 16; off <<= 1) {
            p1 += __shfl_xor(p1, off, 32);
            p2 += __shfl_xor(p2, off, 32);
        }
        if (m == v) {
            const long row = g0 + v + 8 * half;
            s1[row] = p1;
            s2[row] = p2;
        }
    }

    // store hT[b][n][j] as f16 (j within batch)
    const int b  = (int)(g0 >> 11);
    const int jb = (int)(g0 & 2047);
    #pragma unroll
    for (int t = 0; t < 4; ++t) {
        const int n = t * 16 + m;
        #pragma unroll
        for (int v = 0; v < 8; ++v) {
            const int j = jb + v + 8 * half;
            hT[((long)b * 64 + n) * 2048 + j] = (_Float16)acc[t][v];
        }
    }
}

// ---------------------------------------------------------------------------
// Phase 2: flash-style online softmax + h_prime = att @ h.
// One wave = 16 output rows; 8 waves / block share LDS-staged 32x64 h tile.
// Staging via Tensor Data Mover (tensor_load_to_lds + s_wait_tensorcnt),
// with the TDM pad feature producing the 40-half padded row stride.
// ---------------------------------------------------------------------------
__global__ __launch_bounds__(256) void gat_phase2(
    const float* __restrict__ adj, const _Float16* __restrict__ hT,
    const float* __restrict__ s1, const float* __restrict__ s2,
    float* __restrict__ out)
{
    // [n][k] tile, stride 40 halves (80B rows: 16B-aligned, banks 20n mod 64
    // distinct over 16 lanes). Sole __shared__ object -> LDS offset 0 for D#.
    __shared__ _Float16 ldsH[64 * 40];

    const int tid  = threadIdx.x;
    const int wave = tid >> 5;
    const int lane = tid & 31;
    const int m    = lane & 15;
    const int half = lane >> 4;

    const int b      = blockIdx.x >> 4;
    const int tile   = blockIdx.x & 15;
    const int i_base = tile * 128 + wave * 16;
    const int i_row  = i_base + m;

    const float  s1v    = s1[b * 2048 + i_row];
    const float* adjRow = adj + (long)i_row * 2048;
    const float* s2b    = s2 + b * 2048;
    const _Float16* hTb = hT + (long)b * 64 * 2048;

    v8f   acc[4] = {};
    float m_run  = -3.0e38f;
    float l_run  = 0.f;

    for (int jt = 0; jt < 64; ++jt) {
        const int j0 = jt * 32;

        __syncthreads();                            // previous tile reads done
#if __has_builtin(__builtin_amdgcn_tensor_load_to_lds)
        if (wave == 0) {
            // Tensor DMA descriptor (D#) per CDNA5 ISA:
            // 2-D tensor hT[b] (64 x 2048 f16), tile 64 rows x 32 cols,
            // LDS pad: 4 DWORDs after every 16 DWORDs -> 40-half row stride.
            const unsigned long long gaddr =
                (unsigned long long)(uintptr_t)hTb + (unsigned long long)j0 * 2ull;
            v4u g0;
            g0[0] = 1u;                                   // count=1
            g0[1] = 0u;                                   // lds_addr = 0
            g0[2] = (unsigned)(gaddr & 0xFFFFFFFFull);    // global_addr[31:0]
            g0[3] = (unsigned)((gaddr >> 32) & 0x1FFFFFFull) | (2u << 30); // type=2
            v8i g1;
            g1[0] = (int)((1u << 16)      // data_size = 2 bytes
                        | (1u << 20)      // pad_enable
                        | (3u << 22)      // pad_interval: 16 DWORDs
                        | (3u << 25));    // pad_amount:   4 DWORDs
            g1[1] = (int)(2048u << 16);   // tensor_dim0 = 2048 (bits 79:48)
            g1[2] = (int)(64u << 16);     // tensor_dim1 = 64   (bits 111:80)
            g1[3] = (int)(32u << 16);     // tile_dim0   = 32   (bits 127:112)
            g1[4] = (int)(64u);           // tile_dim1   = 64, tile_dim2 = 0
            g1[5] = (int)(2048);          // tensor_dim0_stride = 2048
            g1[6] = 0;
            g1[7] = 0;
            v4i gz = {0, 0, 0, 0};
#if defined(__clang_major__) && (__clang_major__ >= 23)
            v8i gz8 = {0, 0, 0, 0, 0, 0, 0, 0};
            __builtin_amdgcn_tensor_load_to_lds(g0, g1, gz, gz, gz8, 0);
#else
            __builtin_amdgcn_tensor_load_to_lds(g0, g1, gz, gz, 0);
#endif
            __builtin_amdgcn_s_wait_tensorcnt(0);
        }
#else
        {   // fallback: cooperative vector staging
            const int sn = tid >> 2, ssub = tid & 3;
            const _Float16* src = hTb + sn * 2048 + j0 + ssub * 8;
            uint4 v = *(const uint4*)src;
            *(uint4*)&ldsH[sn * 40 + ssub * 8] = v;
        }
#endif
        __syncthreads();

        if (jt < 63) {                               // prefetch next adj tile
            __builtin_prefetch(adjRow + j0 + 32, 0, 0);
        }

        // scores in A-fragment order: two contiguous 8-col runs per lane
        float e[16];
        float tmax = -3.0e38f;
        #pragma unroll
        for (int seg = 0; seg < 2; ++seg) {
            const int kb = j0 + 16 * seg + 8 * half;
            float4 a0 = *(const float4*)(adjRow + kb);
            float4 a1 = *(const float4*)(adjRow + kb + 4);
            float4 v0 = *(const float4*)(s2b + kb);
            float4 v1 = *(const float4*)(s2b + kb + 4);
            const float adjv[8] = {a0.x, a0.y, a0.z, a0.w, a1.x, a1.y, a1.z, a1.w};
            const float s2v[8]  = {v0.x, v0.y, v0.z, v0.w, v1.x, v1.y, v1.z, v1.w};
            #pragma unroll
            for (int off = 0; off < 8; ++off) {
                float ev = s1v + s2v[off];
                ev = ev > 0.f ? ev : ev * ALPHA_SLOPE;   // leaky BEFORE mask
                const bool c = (adjv[off] > 0.f) || (kb + off == i_row); // +eye
                const float val = c ? ev : NEG_INF_F;
                e[8 * seg + off] = val;
                tmax = fmaxf(tmax, val);
            }
        }
        tmax = fmaxf(tmax, __shfl_xor(tmax, 16, 32));
        const float m_new = fmaxf(m_run, tmax);
        const float alpha = __expf(m_run - m_new);

        v16h  af;
        float psum = 0.f;
        #pragma unroll
        for (int q = 0; q < 16; ++q) {
            const float p = __expf(e[q] - m_new);
            psum += p;
            af[q] = (_Float16)p;
        }
        psum += __shfl_xor(psum, 16, 32);
        l_run = l_run * alpha + psum;
        m_run = m_new;

        // rescale accumulators: row of acc[t][v] = v + 8*half
        #pragma unroll
        for (int v = 0; v < 8; ++v) {
            const float av = __shfl(alpha, v + 8 * half, 32);
            #pragma unroll
            for (int t = 0; t < 4; ++t) acc[t][v] *= av;
        }

        // acc += p(16x32) @ h(32x64)
        #pragma unroll
        for (int t = 0; t < 4; ++t) {
            const int n = t * 16 + m;
            const int base = n * 40 + 8 * half;      // two 8-half runs: +0, +16
            v8h lo = *(const v8h*)&ldsH[base];
            v8h hi = *(const v8h*)&ldsH[base + 16];
            v16h bf = __builtin_shufflevector(lo, hi,
                0,1,2,3,4,5,6,7,8,9,10,11,12,13,14,15);
            acc[t] = __builtin_amdgcn_wmma_f32_16x16x32_f16(
                false, af, false, bf, (short)0, acc[t], false, false);
        }
    }

    // normalize and write: out[b][i_base + v + 8*half][t*16 + m]
    float lv[8];
    #pragma unroll
    for (int v = 0; v < 8; ++v) lv[v] = __shfl(l_run, v + 8 * half, 32);

    float* outb = out + ((long)b * 2048 + i_base) * 64;
    #pragma unroll
    for (int t = 0; t < 4; ++t) {
        const int n = t * 16 + m;
        #pragma unroll
        for (int v = 0; v < 8; ++v) {
            outb[(v + 8 * half) * 64 + n] = acc[t][v] / lv[v];
        }
    }
}

// ---------------------------------------------------------------------------
extern "C" void kernel_launch(void* const* d_in, const int* in_sizes, int n_in,
                              void* d_out, int out_size, void* d_ws, size_t ws_size,
                              hipStream_t stream)
{
    (void)in_sizes; (void)n_in; (void)out_size; (void)ws_size;
    const float* inp = (const float*)d_in[0];   // (8,2048,128)
    const float* adj = (const float*)d_in[1];   // (2048,2048)
    const float* W   = (const float*)d_in[2];   // (128,64)
    const float* a   = (const float*)d_in[3];   // (128,1)
    float*       out = (float*)d_out;           // (8,2048,64)

    _Float16* hT = (_Float16*)d_ws;                               // 2 MB
    float*    s1 = (float*)((char*)d_ws + (size_t)8 * 64 * 2048 * 2);
    float*    s2 = s1 + 8 * 2048;

    hipLaunchKernelGGL(gat_phase1, dim3(128), dim3(256), 0, stream,
                       inp, W, a, hT, s1, s2);
    hipLaunchKernelGGL(gat_phase2, dim3(128), dim3(256), 0, stream,
                       adj, hT, s1, s2, out);
}
